// FeatureRNN_53970559042267
// MI455X (gfx1250) — matrix operand order repaired and checked by reading, not compiled, not run on hardware
//
#include <hip/hip_runtime.h>

// ---------------------------------------------------------------------------
// FeatureRNN tree-MLP sweep for MI455X (gfx1250, wave32, WMMA).
// 5 dependent passes; each pass: x[R x 177] -> MLP(177->125->40) -> (sum4).
// f16 WMMA 16x16x32, f32 accumulate. Weights LDS-resident in B-fragment
// layout (2x ds_load_b128 per fragment). Per-kc batched B loads -> one
// dscnt wait -> 8 back-to-back v_wmma. Wave-private tiles, no in-loop
// workgroup barriers (wavefront fences only). f16 intermediates.
// ---------------------------------------------------------------------------

typedef __attribute__((ext_vector_type(16))) _Float16 v16h;
typedef __attribute__((ext_vector_type(8)))  _Float16 v8h;
typedef __attribute__((ext_vector_type(8)))  float    v8f;

#define ATOM 125
#define BOND 12
#define OUTF 40
#define HIDF 125
#define K1P  192   // 177 padded to 6*32
#define N1P  128   // 125 padded to 8*16
#define K2P  128   // 125 padded to 4*32
#define N2P  48    // 40  padded to 3*16

__device__ __forceinline__ v16h hcat(v8h lo, v8h hi) {
  return __builtin_shufflevector(lo, hi, 0,1,2,3,4,5,6,7,8,9,10,11,12,13,14,15);
}
__device__ __forceinline__ float lrelu(float v) { return v > 0.f ? v : 0.01f * v; }
// Wave-private LDS phase boundary: DS ops from one wave complete in order, we
// only need to stop the compiler from reordering across the phase.
__device__ __forceinline__ void wavefence() {
  __builtin_amdgcn_fence(__ATOMIC_ACQ_REL, "wavefront");
}

__global__ __launch_bounds__(128)
void frnn_level_kernel(const float*    __restrict__ msgs,   // level msgs, x125 (row>>msg_shift)
                       const float*    __restrict__ bondp,  // rows x 12 or nullptr
                       const _Float16* __restrict__ childh, // rows x 40 (f16) or nullptr
                       const float* __restrict__ W1, const float* __restrict__ b1,
                       const float* __restrict__ W2, const float* __restrict__ b2,
                       float*    __restrict__ outf,         // f32 out (final pass) or nullptr
                       _Float16* __restrict__ outh,         // f16 out (intermediate) or nullptr
                       int R, int msg_shift, int reduce4)
{
  // --- LDS -----------------------------------------------------------------
  __shared__ _Float16 w1f[6 * 8 * 32 * 16];   // B-fragment swizzled W1 (48 KB)
  __shared__ _Float16 w2f[4 * 3 * 32 * 16];   // B-fragment swizzled W2 (12 KB)
  __shared__ float    b1s[N1P];
  __shared__ float    b2s[N2P];
  __shared__ _Float16 xs[4][16 * K1P];        // wave-private activation tiles (24 KB)
  __shared__ _Float16 hs[4][16 * K2P];        // wave-private hidden tiles (16 KB)

  const int tid  = threadIdx.x;
  const int lane = tid & 31;
  const int wave = tid >> 5;
  const int arow = lane & 15;   // A-row / B,C,D-col owned by this lane
  const int asel = lane >> 4;   // K-half (A/B) or M-half (C/D)

  // --- Stage weights once, swizzled to B-fragment layout -------------------
  // B frag order chosen here (contiguous k, same as A): lane<16 -> N=lane, halves j = K 0..15;
  //                         lane>=16 -> N=lane-16, halves j = K 16..31.
  for (int i = tid; i < 6 * 8 * 32 * 16; i += 128) {
    int j  = i & 15;
    int ls = (i >> 4) & 31;
    int nt = (i >> 9) & 7;
    int kc = i >> 12;
    int k  = kc * 32 + ((ls >> 4) << 4) + j;
    int n  = nt * 16 + (ls & 15);
    float v = (k < 177 && n < HIDF) ? W1[k * HIDF + n] : 0.f;
    w1f[i] = (_Float16)v;
  }
  for (int i = tid; i < 4 * 3 * 32 * 16; i += 128) {
    int j  = i & 15;
    int ls = (i >> 4) & 31;
    int r  = i >> 9;           // 0..11
    int nt = r % 3;
    int kc = r / 3;
    int k  = kc * 32 + ((ls >> 4) << 4) + j;
    int n  = nt * 16 + (ls & 15);
    float v = (k < HIDF && n < OUTF) ? W2[k * OUTF + n] : 0.f;
    w2f[i] = (_Float16)v;
  }
  for (int i = tid; i < N1P; i += 128) b1s[i] = (i < HIDF) ? b1[i] : 0.f;
  for (int i = tid; i < N2P; i += 128) b2s[i] = (i < OUTF) ? b2[i] : 0.f;
  __syncthreads();   // weights visible to all waves; the ONLY workgroup barrier

  _Float16* xw = xs[wave];
  _Float16* hw = hs[wave];

  // Waves are fully decoupled: each wave owns a 16-row tile; stride = total waves.
  const int ntilesW = (R + 15) >> 4;                      // 16-row wave tiles
  const int wgid    = blockIdx.x * 4 + wave;
  const int wstride = gridDim.x * 4;

  for (int wt = wgid; wt < ntilesW; wt += wstride) {
    const int row0 = wt * 16;

    // --- Phase 1: gather 16 rows of x[177] -> f16, zero-padded to 192 ------
    for (int e = lane; e < 16 * K1P; e += 32) {
      int lr = e / K1P;
      int k  = e - lr * K1P;
      int rg = row0 + lr;
      float v = 0.f;
      if (rg < R) {
        if (k < ATOM) {
          v = msgs[(size_t)(rg >> msg_shift) * ATOM + k];
        } else if (k < ATOM + BOND) {
          if (bondp) v = bondp[(size_t)rg * BOND + (k - ATOM)];
        } else if (k < ATOM + BOND + OUTF) {
          if (childh) v = (float)childh[(size_t)rg * OUTF + (k - (ATOM + BOND))];
        }
      }
      xw[e] = (_Float16)v;
    }
    wavefence();

    // --- Phase 2: GEMM1  (16x192) x (192x128) -> acc1[8] -------------------
    v8f acc1[8];
#pragma unroll
    for (int nt = 0; nt < 8; ++nt) acc1[nt] = v8f{0.f,0.f,0.f,0.f,0.f,0.f,0.f,0.f};
#pragma unroll
    for (int kc = 0; kc < 6; ++kc) {
      // Batch: issue all fragment loads for this K-chunk, then burst 8 WMMAs.
      v8h blo[8], bhi[8];
#pragma unroll
      for (int nt = 0; nt < 8; ++nt) {
        const v8h* bp = (const v8h*)(w1f + (((kc * 8 + nt) * 32 + lane) << 4));
        blo[nt] = bp[0];
        bhi[nt] = bp[1];
      }
      const v8h* ap = (const v8h*)(xw + arow * K1P + kc * 32 + asel * 16);
      v16h a = hcat(ap[0], ap[1]);   // +8 halves = +1 v8h
#pragma unroll
      for (int nt = 0; nt < 8; ++nt) {
        acc1[nt] = __builtin_amdgcn_wmma_f32_16x16x32_f16(
            false, a, false, hcat(blo[nt], bhi[nt]), (short)0, acc1[nt], false, false);
      }
    }

    // --- Phase 3: bias + leaky-relu -> h (f16 in LDS, zero-padded cols) ----
    // C/D layout: lane holds (M = v + 8*asel, N = nt*16 + arow) for v=0..7
#pragma unroll
    for (int nt = 0; nt < 8; ++nt) {
      int col = nt * 16 + arow;
      float bb = b1s[col];           // 0 for padded cols -> h stays 0 there
#pragma unroll
      for (int v = 0; v < 8; ++v) {
        int r = v + (asel << 3);
        hw[r * K2P + col] = (_Float16)lrelu(acc1[nt][v] + bb);
      }
    }
    wavefence();

    // --- Phase 4: GEMM2  (16x128) x (128x48) -> acc2[3] --------------------
    v8f acc2[3];
#pragma unroll
    for (int nt = 0; nt < 3; ++nt) acc2[nt] = v8f{0.f,0.f,0.f,0.f,0.f,0.f,0.f,0.f};
#pragma unroll
    for (int kc = 0; kc < 4; ++kc) {
      v8h blo[3], bhi[3];
#pragma unroll
      for (int nt = 0; nt < 3; ++nt) {
        const v8h* bp = (const v8h*)(w2f + (((kc * 3 + nt) * 32 + lane) << 4));
        blo[nt] = bp[0];
        bhi[nt] = bp[1];
      }
      const v8h* ap = (const v8h*)(hw + arow * K2P + kc * 32 + asel * 16);
      v16h a = hcat(ap[0], ap[1]);
#pragma unroll
      for (int nt = 0; nt < 3; ++nt) {
        acc2[nt] = __builtin_amdgcn_wmma_f32_16x16x32_f16(
            false, a, false, hcat(blo[nt], bhi[nt]), (short)0, acc2[nt], false, false);
      }
    }

    // --- Phase 5: epilogue --------------------------------------------------
    if (reduce4) {
      // rows 4c..4c+3 of a parent live in this lane's acc VGPRs -> free sum
      int pbase = row0 >> 2;
      int nP    = R >> 2;
#pragma unroll
      for (int nt = 0; nt < 3; ++nt) {
        int col = nt * 16 + arow;
        if (col < OUTF) {
          float bb = b2s[col];
          float s0 = 0.f, s1 = 0.f;
#pragma unroll
          for (int v = 0; v < 4; ++v) s0 += lrelu(acc2[nt][v] + bb);
#pragma unroll
          for (int v = 4; v < 8; ++v) s1 += lrelu(acc2[nt][v] + bb);
          int p0 = pbase + asel * 2;
          int p1 = p0 + 1;
          if (outh) {
            if (p0 < nP) outh[(size_t)p0 * OUTF + col] = (_Float16)s0;
            if (p1 < nP) outh[(size_t)p1 * OUTF + col] = (_Float16)s1;
          } else {
            if (p0 < nP) outf[(size_t)p0 * OUTF + col] = s0;
            if (p1 < nP) outf[(size_t)p1 * OUTF + col] = s1;
          }
        }
      }
    } else {
#pragma unroll
      for (int nt = 0; nt < 3; ++nt) {
        int col = nt * 16 + arow;
        if (col < OUTF) {
          float bb = b2s[col];
#pragma unroll
          for (int v = 0; v < 8; ++v) {
            int rg = row0 + v + (asel << 3);
            if (rg < R) outh[(size_t)rg * OUTF + col] = (_Float16)lrelu(acc2[nt][v] + bb);
          }
        }
      }
    }
    wavefence();   // next iteration overwrites xs/hs
  }
}

// ---------------------------------------------------------------------------
// Host-side launch: 5 sequential passes on `stream`, f16 ping-pong in d_ws.
// ---------------------------------------------------------------------------
extern "C" void kernel_launch(void* const* d_in, const int* in_sizes, int n_in,
                              void* d_out, int out_size, void* d_ws, size_t ws_size,
                              hipStream_t stream) {
  (void)in_sizes; (void)n_in; (void)out_size; (void)ws_size;

  const float* node_msg = (const float*)d_in[0];
  const float* bonds    = (const float*)d_in[1];
  const float* W1       = (const float*)d_in[2];
  const float* b1       = (const float*)d_in[3];
  const float* W2       = (const float*)d_in[4];
  const float* b2       = (const float*)d_in[5];
  float*       out      = (float*)d_out;

  // SIZES = {2048, 8192, 32768, 131072, 524288}
  static const long long NOFF[6]  = {0, 2048, 10240, 43008, 174080, 698368};
  static const long long BOFF[5]  = {0, 8192, 40960, 172032, 696320};
  static const int       RROWS[5] = {8192, 32768, 131072, 524288, 524288}; // rows, d=0..4

  _Float16* bufA = (_Float16*)d_ws;                  // up to 524288 x 40 f16 (~42 MB)
  _Float16* bufB = bufA + (size_t)524288 * OUTF;     // up to 131072 x 40 f16 (~10.5 MB)

  const int block = 128;
  auto grid_for = [](int R) {
    int nt = (R + 63) >> 6;          // 64 rows per workgroup stride unit
    return (nt < 2048) ? nt : 2048;
  };

  // Pass 0: leaves (d=4 slice of node_msg) -> bufA (f16), no reduce.
  frnn_level_kernel<<<grid_for(RROWS[4]), block, 0, stream>>>(
      node_msg + NOFF[4] * ATOM, nullptr, nullptr,
      W1, b1, W2, b2, nullptr, bufA, RROWS[4], /*msg_shift=*/0, /*reduce4=*/0);

  // Pass 1: d=3   child=bufA -> bufB
  frnn_level_kernel<<<grid_for(RROWS[3]), block, 0, stream>>>(
      node_msg + NOFF[3] * ATOM, bonds + BOFF[3] * BOND, bufA,
      W1, b1, W2, b2, nullptr, bufB, RROWS[3], 2, 1);

  // Pass 2: d=2   child=bufB -> bufA
  frnn_level_kernel<<<grid_for(RROWS[2]), block, 0, stream>>>(
      node_msg + NOFF[2] * ATOM, bonds + BOFF[2] * BOND, bufB,
      W1, b1, W2, b2, nullptr, bufA, RROWS[2], 2, 1);

  // Pass 3: d=1   child=bufA -> bufB
  frnn_level_kernel<<<grid_for(RROWS[1]), block, 0, stream>>>(
      node_msg + NOFF[1] * ATOM, bonds + BOFF[1] * BOND, bufA,
      W1, b1, W2, b2, nullptr, bufB, RROWS[1], 2, 1);

  // Pass 4: d=0   child=bufB -> d_out (f32, 2048 x 40)
  frnn_level_kernel<<<grid_for(RROWS[0]), block, 0, stream>>>(
      node_msg + NOFF[0] * ATOM, bonds + BOFF[0] * BOND, bufB,
      W1, b1, W2, b2, out, nullptr, RROWS[0], 2, 1);
}
